// AutoregressiveLSTM_9534827397434
// MI455X (gfx1250) — compile-verified, matrix operand order checked by reading, not run on hardware
//
#include <hip/hip_runtime.h>
#include <hip/hip_bf16.h>

// ---------------------------------------------------------------------------
// AutoregressiveLSTM on MI455X (gfx1250): WMMA f16 16x16x32, f32 accumulate.
// B=512, T=256, I=64, H=512, STEPS=10.
// ---------------------------------------------------------------------------

typedef __attribute__((ext_vector_type(16))) _Float16 v16h;
typedef __attribute__((ext_vector_type(8)))  float    v8f;
typedef _Float16 half_t;

#define BB   512   // batch
#define TT   256   // time steps
#define II   64    // input size
#define HH   512   // hidden size
#define G4H  2048  // 4*H
#define NSTEP 10

union F16Frag { v16h v; float4 f[2]; };
union H8      { float4 f4; half_t h[8]; };

__device__ __forceinline__ float sigf(float x) { return 1.0f / (1.0f + __expf(-x)); }

// ---------------------------------------------------------------------------
// Pack weights into WMMA B-fragment order (f16).
// B matrix is K x N with B[k][n] = W[n][k];  W is (N x K) row-major fp32.
// Combined K = K0 (input weights) + K1 (hidden weights).
// Fragment (kt, ntile): 32 lanes x 16 halves contiguous.
// Lane layout (mirrors 16-bit A/B WMMA layout): n = ntile*16 + (lane&15);
// lanes 0-15 hold K = {0..7, 16..23}, lanes 16-31 hold K = {8..15, 24..31}.
// ---------------------------------------------------------------------------
__global__ void pack_b_kernel(half_t* __restrict__ Bsw,
                              const float* __restrict__ Wih, int K0,
                              const float* __restrict__ Whh, int K1,
                              int NT, int KT) {
  int tid = blockIdx.x * blockDim.x + threadIdx.x;
  int total = KT * NT * 32;
  if (tid >= total) return;
  int lane  = tid & 31;
  int fi    = tid >> 5;
  int ntile = fi % NT;
  int kt    = fi / NT;
  int n     = ntile * 16 + (lane & 15);
  int kb    = (lane & 16) ? 8 : 0;
  half_t* o = Bsw + (size_t)tid * 16;
#pragma unroll
  for (int e = 0; e < 16; ++e) {
    int klocal = (e < 8) ? (kb + e) : (kb + 8 + e); // kb+16+(e-8)
    int kg = kt * 32 + klocal;
    float v;
    if (kg < K0) v = Wih[(size_t)n * K0 + kg];
    else         v = Whh[(size_t)n * K1 + (kg - K0)];
    o[e] = (half_t)v;
  }
}

__global__ void add2_kernel(float* __restrict__ o, const float* __restrict__ a,
                            const float* __restrict__ b, int n) {
  int i = blockIdx.x * blockDim.x + threadIdx.x;
  if (i < n) o[i] = a[i] + (b ? b[i] : 0.0f);
}

__global__ void zero_f16_kernel(half_t* p, int n) {
  int i = blockIdx.x * blockDim.x + threadIdx.x;
  if (i < n) p[i] = (half_t)0.0f;
}
__global__ void zero_f32_kernel(float* p, int n) {
  int i = blockIdx.x * blockDim.x + threadIdx.x;
  if (i < n) p[i] = 0.0f;
}

// ---------------------------------------------------------------------------
// Fused LSTM cell step:
//   gates[B,4H] = [A0 | A1] @ Bsw  + bias     (combined K = k0 + 512)
//   c_new = sig(f)*c + sig(i)*tanh(g);  h_new = sig(o)*tanh(c_new)
// A0: either fp32 (x slice / decoder input, row stride lda0) or f16.
// A1: previous hidden state, f16, compact B x H.
// Grid (8,8): block = 64 batch rows x 64 hidden cols x all 4 gates.
// 8 waves: wm(2) x wn(4); each wave: 32 rows x 16 cols x 4 gates =
// 8 f32 WMMA accumulators -> full cell update stays in registers.
// ---------------------------------------------------------------------------
template <bool A0F32>
__global__ __launch_bounds__(256)
void lstm_step_kernel(const void* __restrict__ A0_, int lda0, int k0,
                      const half_t* __restrict__ A1,
                      const half_t* __restrict__ Bsw,
                      const float* __restrict__ bias,
                      float* __restrict__ c_io,
                      half_t* __restrict__ h_out) {
  constexpr int NT = G4H / 16; // 128 n-tiles across 4 gates
  const int KT = (k0 + HH) >> 5;

  __shared__ __align__(16) half_t Atile[64 * 40]; // 64 rows x 32 halves, pad 8

  const int tid  = threadIdx.x;
  const int lane = tid & 31;
  const int wave = tid >> 5;
  const int wm   = wave >> 2; // 0..1  -> 32-row half
  const int wn   = wave & 3;  // 0..3  -> 16-col slice
  const int mBlock = blockIdx.x * 64;
  const int nBlock = blockIdx.y * 64;

  v8f acc[4][2];
#pragma unroll
  for (int g = 0; g < 4; ++g)
#pragma unroll
    for (int mt = 0; mt < 2; ++mt)
#pragma unroll
      for (int j = 0; j < 8; ++j) acc[g][mt][j] = 0.0f;

  // cooperative A-tile staging: 256 threads, each moves 8 halves (16B)
  const int sr = tid >> 2;          // 0..63 row in tile
  const int sc = (tid & 3) * 8;     // 0,8,16,24 k-chunk
  const size_t aRow = (size_t)(mBlock + sr);
  const int kbA = (lane & 16) ? 8 : 0;
  const int ntBase = (nBlock >> 4) + wn; // + g*32 per gate

  for (int kt = 0; kt < KT; ++kt) {
    const int kk = kt * 32 + sc;
    __syncthreads();
    {
      H8 t;
      if (kk < k0) {
        if (A0F32) {
          const float* p = (const float*)A0_ + aRow * (size_t)lda0 + kk;
#pragma unroll
          for (int j = 0; j < 8; ++j) t.h[j] = (half_t)p[j];
        } else {
          const half_t* p = (const half_t*)A0_ + aRow * (size_t)lda0 + kk;
          t.f4 = *(const float4*)p;
        }
      } else {
        const half_t* p = A1 + aRow * HH + (kk - k0);
        t.f4 = *(const float4*)p;
      }
      *(float4*)&Atile[sr * 40 + sc] = t.f4;
    }
    __syncthreads();

    // A fragments (2 m-tiles) from LDS
    F16Frag afr[2];
#pragma unroll
    for (int mt = 0; mt < 2; ++mt) {
      const int lr = wm * 32 + mt * 16 + (lane & 15);
      afr[mt].f[0] = *(const float4*)&Atile[lr * 40 + kbA];
      afr[mt].f[1] = *(const float4*)&Atile[lr * 40 + kbA + 16];
    }

    // B fragments: one per gate, direct from pre-swizzled weights (L2-resident)
#pragma unroll
    for (int g = 0; g < 4; ++g) {
      const half_t* bp =
          Bsw + (((size_t)kt * NT + (g * 32 + ntBase)) * 32 + lane) * 16;
      v16h bfr = *(const v16h*)bp;
      if (kt + 1 < KT)
        __builtin_prefetch(bp + (size_t)NT * 32 * 16, 0, 1); // global_prefetch_b8
#pragma unroll
      for (int mt = 0; mt < 2; ++mt)
        acc[g][mt] = __builtin_amdgcn_wmma_f32_16x16x32_f16(
            false, afr[mt].v, false, bfr, (short)0, acc[g][mt], false, false);
    }
  }

  // Epilogue: fused LSTM cell, all 4 gates in-register per (row,col)
  const int col   = nBlock + wn * 16 + (lane & 15);
  const int rbase = mBlock + wm * 32 + ((lane & 16) ? 8 : 0);
  const float bi = bias[col];
  const float bf = bias[HH + col];
  const float bg = bias[2 * HH + col];
  const float bo = bias[3 * HH + col];
#pragma unroll
  for (int mt = 0; mt < 2; ++mt) {
#pragma unroll
    for (int j = 0; j < 8; ++j) {
      const int row = rbase + mt * 16 + j;
      const size_t idx = (size_t)row * HH + col;
      const float iv = acc[0][mt][j] + bi;
      const float fv = acc[1][mt][j] + bf;
      const float gv = acc[2][mt][j] + bg;
      const float ov = acc[3][mt][j] + bo;
      const float cp = c_io[idx];
      const float cn = sigf(fv) * cp + sigf(iv) * tanhf(gv);
      c_io[idx] = cn;
      h_out[idx] = (half_t)(sigf(ov) * tanhf(cn));
    }
  }
}

// ---------------------------------------------------------------------------
// Decoder output projection: out[B,64] = h1 @ W_lin^T + b_lin
// M=512, N=64, K=512. Grid 32 blocks x 4 waves; one 16x16 tile per wave.
// Writes strided d_out (b, s, i) AND compact f32 "cur" for the next step.
// ---------------------------------------------------------------------------
__global__ __launch_bounds__(128)
void out_proj_kernel(const half_t* __restrict__ A,    // B x H, f16
                     const half_t* __restrict__ BswL, // swizzled 512x64
                     const float* __restrict__ biasL,
                     float* __restrict__ outStrided, int ostride,
                     float* __restrict__ cur) {
  const int lane = threadIdx.x & 31;
  const int wn   = threadIdx.x >> 5; // n-tile 0..3
  const int mBase = blockIdx.x * 16;
  const int kbA = (lane & 16) ? 8 : 0;
  const int row = mBase + (lane & 15);

  v8f acc;
#pragma unroll
  for (int j = 0; j < 8; ++j) acc[j] = 0.0f;

  for (int kt = 0; kt < HH / 32; ++kt) {
    F16Frag a;
    const half_t* ap = A + (size_t)row * HH + kt * 32 + kbA;
    a.f[0] = *(const float4*)ap;
    a.f[1] = *(const float4*)(ap + 16);
    const half_t* bp = BswL + (((size_t)kt * 4 + wn) * 32 + lane) * 16;
    v16h b = *(const v16h*)bp;
    acc = __builtin_amdgcn_wmma_f32_16x16x32_f16(false, a.v, false, b,
                                                 (short)0, acc, false, false);
  }

  const int col = wn * 16 + (lane & 15);
  const int rb  = mBase + ((lane & 16) ? 8 : 0);
  const float bv = biasL[col];
#pragma unroll
  for (int j = 0; j < 8; ++j) {
    const int r = rb + j;
    const float v = acc[j] + bv;
    outStrided[(size_t)r * ostride + col] = v;
    cur[(size_t)r * II + col] = v;
  }
}

// ---------------------------------------------------------------------------
extern "C" void kernel_launch(void* const* d_in, const int* in_sizes, int n_in,
                              void* d_out, int out_size, void* d_ws, size_t ws_size,
                              hipStream_t stream) {
  const float* x     = (const float*)d_in[0];  // (B, T, I)
  const float* W_ih0 = (const float*)d_in[1];  // (4H, I)
  const float* W_hh0 = (const float*)d_in[2];  // (4H, H)
  const float* b_ih0 = (const float*)d_in[3];
  const float* b_hh0 = (const float*)d_in[4];
  const float* W_ih1 = (const float*)d_in[5];  // (4H, H)
  const float* W_hh1 = (const float*)d_in[6];  // (4H, H)
  const float* b_ih1 = (const float*)d_in[7];
  const float* b_hh1 = (const float*)d_in[8];
  const float* W_lin = (const float*)d_in[9];  // (I, H)
  const float* b_lin = (const float*)d_in[10];
  float* out = (float*)d_out;                  // (B, STEPS, I)

  // ---- workspace carve (256B aligned) ----
  char* w = (char*)d_ws;
  auto carve = [&](size_t bytes) -> void* {
    void* p = (void*)w;
    w += (bytes + 255) & ~(size_t)255;
    return p;
  };
  half_t* Bsw0 = (half_t*)carve((size_t)(II + HH) * G4H * sizeof(half_t)); // 576 x 2048
  half_t* Bsw1 = (half_t*)carve((size_t)(HH + HH) * G4H * sizeof(half_t)); // 1024 x 2048
  half_t* BswL = (half_t*)carve((size_t)HH * II * sizeof(half_t));         // 512 x 64
  float* bias0 = (float*)carve(G4H * sizeof(float));
  float* bias1 = (float*)carve(G4H * sizeof(float));
  float* biasL = (float*)carve(II * sizeof(float));
  half_t* h0a  = (half_t*)carve((size_t)BB * HH * sizeof(half_t));
  half_t* h0b  = (half_t*)carve((size_t)BB * HH * sizeof(half_t));
  half_t* h1a  = (half_t*)carve((size_t)BB * HH * sizeof(half_t));
  half_t* h1b  = (half_t*)carve((size_t)BB * HH * sizeof(half_t));
  float* c0    = (float*)carve((size_t)BB * HH * sizeof(float));
  float* c1    = (float*)carve((size_t)BB * HH * sizeof(float));
  float* cur   = (float*)carve((size_t)BB * II * sizeof(float));

  // ---- pack weights + biases + zero state ----
  {
    int kt0 = (II + HH) / 32, kt1 = (HH + HH) / 32, ktL = HH / 32;
    int n0 = kt0 * 128 * 32, n1 = kt1 * 128 * 32, nL = ktL * 4 * 32;
    pack_b_kernel<<<(n0 + 255) / 256, 256, 0, stream>>>(Bsw0, W_ih0, II, W_hh0, HH, 128, kt0);
    pack_b_kernel<<<(n1 + 255) / 256, 256, 0, stream>>>(Bsw1, W_ih1, HH, W_hh1, HH, 128, kt1);
    pack_b_kernel<<<(nL + 255) / 256, 256, 0, stream>>>(BswL, W_lin, HH, nullptr, 0, 4, ktL);
    add2_kernel<<<(G4H + 255) / 256, 256, 0, stream>>>(bias0, b_ih0, b_hh0, G4H);
    add2_kernel<<<(G4H + 255) / 256, 256, 0, stream>>>(bias1, b_ih1, b_hh1, G4H);
    add2_kernel<<<1, 64, 0, stream>>>(biasL, b_lin, nullptr, II);
    int nh = BB * HH;
    zero_f16_kernel<<<(nh + 255) / 256, 256, 0, stream>>>(h0a, nh);
    zero_f16_kernel<<<(nh + 255) / 256, 256, 0, stream>>>(h1a, nh);
    zero_f32_kernel<<<(nh + 255) / 256, 256, 0, stream>>>(c0, nh);
    zero_f32_kernel<<<(nh + 255) / 256, 256, 0, stream>>>(c1, nh);
  }

  dim3 sg(BB / 64, HH / 64); // (8, 8)
  dim3 sb(256);
  half_t *h0in = h0a, *h0out = h0b, *h1in = h1a, *h1out = h1b;

  // ---- encoder: interleave layer0 / layer1 per timestep ----
  for (int t = 0; t < TT; ++t) {
    lstm_step_kernel<true><<<sg, sb, 0, stream>>>(
        (const void*)(x + (size_t)t * II), TT * II, II, h0in, Bsw0, bias0, c0, h0out);
    lstm_step_kernel<false><<<sg, sb, 0, stream>>>(
        (const void*)h0out, HH, HH, h1in, Bsw1, bias1, c1, h1out);
    half_t* tmp;
    tmp = h0in; h0in = h0out; h0out = tmp;
    tmp = h1in; h1in = h1out; h1out = tmp;
  }

  // ---- autoregressive decoder ----
  const float* curp = x + (size_t)(TT - 1) * II; // x[:, -1, :]
  int curlda = TT * II;
  for (int s = 0; s < NSTEP; ++s) {
    lstm_step_kernel<true><<<sg, sb, 0, stream>>>(
        (const void*)curp, curlda, II, h0in, Bsw0, bias0, c0, h0out);
    lstm_step_kernel<false><<<sg, sb, 0, stream>>>(
        (const void*)h0out, HH, HH, h1in, Bsw1, bias1, c1, h1out);
    out_proj_kernel<<<BB / 16, 128, 0, stream>>>(
        h1out, BswL, biasL, out + (size_t)s * II, NSTEP * II, cur);
    curp = cur;
    curlda = II;
    half_t* tmp;
    tmp = h0in; h0in = h0out; h0out = tmp;
    tmp = h1in; h1in = h1out; h1out = tmp;
  }
}
